// Naive_30700426232145
// MI455X (gfx1250) — compile-verified, matrix operand order checked by reading, not run on hardware
//
#include <hip/hip_runtime.h>

// Sizes from the reference.
#define NTOK 16384
#define DIN  2048
#define DOUT 2048
#define NEXP 8

// GEMM tiling.
#define BM 128
#define BN 128
#define BK 32
#define KSTEPS (DIN / BK)  // 64
#define LDSW 40   // padded LDS row stride in halfs (80B) -> conflict-free b128 frag loads
#define NBUF 3    // LDS ring depth (3 buffers -> single barrier per K-step is hazard-free)

typedef _Float16 h4  __attribute__((ext_vector_type(4)));
typedef _Float16 h8  __attribute__((ext_vector_type(8)));
typedef _Float16 h16 __attribute__((ext_vector_type(16)));
typedef float    f4  __attribute__((ext_vector_type(4)));
typedef float    f8  __attribute__((ext_vector_type(8)));

// ---------------- f32 -> f16 conversion (vectorized 4-wide) ----------------
__global__ __launch_bounds__(256) void k_cvt_f32_f16(const float* __restrict__ in,
                                                     _Float16* __restrict__ out,
                                                     int n4) {
  int i = blockIdx.x * 256 + threadIdx.x;
  if (i < n4) {
    f4 v = __builtin_nontemporal_load((const f4*)in + i);  // f32 read once
    h4 h = { (_Float16)v.x, (_Float16)v.y, (_Float16)v.z, (_Float16)v.w };
    ((h4*)out)[i] = h;  // regular store: f16 copies should stay L2-resident
  }
}

// ---------------- routing: histogram -> scan -> slot assignment ----------------
__global__ void k_zero(int* p, int n) {
  int i = blockIdx.x * blockDim.x + threadIdx.x;
  if (i < n) p[i] = 0;
}

__global__ __launch_bounds__(256) void k_hist(const int* __restrict__ idxs,
                                              int* __restrict__ counts) {
  int i = blockIdx.x * 256 + threadIdx.x;
  if (i < NTOK) atomicAdd(&counts[idxs[i]], 1);
}

__global__ void k_scan(const int* __restrict__ counts, int* __restrict__ offsets,
                       int* __restrict__ cursors) {
  if (threadIdx.x == 0 && blockIdx.x == 0) {
    int s = 0;
    for (int e = 0; e < NEXP; ++e) {
      offsets[e] = s;
      cursors[e] = s;
      s += counts[e];
    }
  }
}

__global__ __launch_bounds__(256) void k_assign(const int* __restrict__ idxs,
                                                int* __restrict__ cursors,
                                                int* __restrict__ row_map) {
  int i = blockIdx.x * 256 + threadIdx.x;
  if (i < NTOK) {
    int e = idxs[i];
    int slot = atomicAdd(&cursors[e], 1);
    row_map[slot] = i;
  }
}

// ---------------- CDNA5 async global->LDS copy (16B per lane, ASYNCcnt) ----------------
// GV mode: per-lane 64-bit global address in a VGPR pair, per-lane LDS byte
// address in VDST. Tracked by ASYNCcnt; we wait explicitly with s_wait_asynccnt.
__device__ __forceinline__ void async_cp16(unsigned lds_off, unsigned long long gaddr) {
  asm volatile("global_load_async_to_lds_b128 %0, %1, off"
               :: "v"(lds_off), "v"(gaddr)
               : "memory");
}

// ---------------- WMMA fragment load from LDS (K-major, CDNA5 f16 layout) ----------------
// lane L: row = L%16, K-chunk = (L/16)*8 ; halves [0..7] = K in [kc, kc+8),
// halves [8..15] = K in [16+kc, 16+kc+8). Two ds_load_b128 per fragment.
__device__ __forceinline__ h16 load_frag16(const _Float16* __restrict__ base, int lane) {
  const int r  = lane & 15;
  const int kc = (lane >> 4) << 3;  // 0 or 8
  const _Float16* p = base + r * LDSW;
  h8 lo = *(const h8*)(p + kc);
  h8 hi = *(const h8*)(p + 16 + kc);
  h16 f;
#pragma unroll
  for (int i = 0; i < 8; ++i) { f[i] = lo[i]; f[8 + i] = hi[i]; }
  return f;
}

// ---------------- grouped-expert GEMM: y = relu(gather(x) @ W[e]^T + b[e]) ----------------
__global__ __launch_bounds__(256, 1) void k_moe_gemm(
    const _Float16* __restrict__ X,    // [NTOK, DIN] f16 (original token order)
    const _Float16* __restrict__ Wh,   // [NEXP, DOUT, DIN] f16
    const float* __restrict__ bias,    // [NEXP, DOUT] f32
    const int* __restrict__ row_map,   // [NTOK] slot -> token
    const int* __restrict__ counts,    // [NEXP]
    const int* __restrict__ offsets,   // [NEXP]
    float* __restrict__ out)           // [NTOK, DOUT] f32
{
  const int e       = blockIdx.z;
  const int cnt     = counts[e];
  const int m_block = blockIdx.y * BM;
  if (m_block >= cnt) return;  // tile past this expert's segment
  const int seg     = offsets[e];
  const int n_block = blockIdx.x * BN;

  __shared__ __align__(16) _Float16 sA[NBUF][BM * LDSW];
  __shared__ __align__(16) _Float16 sB[NBUF][BN * LDSW];

  const int tid  = threadIdx.x;
  const int lane = tid & 31;
  const int wave = tid >> 5;
  const int wm   = (wave & 1) * 64;   // wave's row offset in block tile
  const int wn   = (wave >> 1) * 32;  // wave's col offset in block tile

  // Each thread owns 2 A-chunks and 2 B-chunks of 16B per K-step.
  // chunk id c in [0,512): row = c>>2, kc = (c&3)*8 halfs.
  unsigned aoff[2], boff[2];             // LDS byte offsets within one tile buffer
  unsigned long long ga[2], gb[2];       // per-thread global addresses (advance 64B/step)
#pragma unroll
  for (int j = 0; j < 2; ++j) {
    int c   = tid + j * 256;
    int row = c >> 2;
    int kc  = (c & 3) * 8;
    aoff[j] = (unsigned)(row * LDSW + kc) * 2u;
    boff[j] = aoff[j];
    int slot = seg + m_block + row;          // rows past cnt read a clamped valid
    if (slot > NTOK - 1) slot = NTOK - 1;    // row; their outputs are masked off.
    int orig = row_map[slot];
    ga[j] = (unsigned long long)(const void*)(X + (size_t)orig * DIN + kc);
    gb[j] = (unsigned long long)(const void*)(Wh + ((size_t)e * DOUT + n_block + row) * DIN + kc);
  }

  const unsigned sAbase = (unsigned)(unsigned long long)(const void*)&sA[0][0];
  const unsigned sBbase = (unsigned)(unsigned long long)(const void*)&sB[0][0];
  const unsigned tileBytes = (unsigned)(BM * LDSW * 2);  // 10240 per tile buffer

  const f8 fzero = {0.f, 0.f, 0.f, 0.f, 0.f, 0.f, 0.f, 0.f};
  f8 acc[4][2];
#pragma unroll
  for (int mt = 0; mt < 4; ++mt)
#pragma unroll
    for (int nt = 0; nt < 2; ++nt) acc[mt][nt] = fzero;

  // Issue one stage: 4 async b128 copies per thread (2 A, 2 B), then advance K.
  auto issue_stage = [&](int buf) {
    unsigned ab = sAbase + (unsigned)buf * tileBytes;
    unsigned bb = sBbase + (unsigned)buf * tileBytes;
#pragma unroll
    for (int j = 0; j < 2; ++j) async_cp16(ab + aoff[j], ga[j]);
#pragma unroll
    for (int j = 0; j < 2; ++j) async_cp16(bb + boff[j], gb[j]);
#pragma unroll
    for (int j = 0; j < 2; ++j) { ga[j] += BK * 2; gb[j] += BK * 2; }
  };

  issue_stage(0);  // prologue: stage 0 -> buf 0

  int buf = 0, nbuf = 1;
  for (int k = 0; k < KSTEPS; ++k) {
    if (k < KSTEPS - 1) {
      issue_stage(nbuf);  // prefetch stage k+1 into next ring slot
      asm volatile("s_wait_asynccnt 0x4" ::: "memory");  // stage k complete (in-order)
    } else {
      asm volatile("s_wait_asynccnt 0x0" ::: "memory");
    }
    __syncthreads();  // single barrier per step: 3-buffer ring makes this sufficient

    const _Float16* bufA = &sA[buf][0];
    const _Float16* bufB = &sB[buf][0];
    h16 afr[4], bfr[2];
#pragma unroll
    for (int mt = 0; mt < 4; ++mt)
      afr[mt] = load_frag16(bufA + (wm + mt * 16) * LDSW, lane);
#pragma unroll
    for (int nt = 0; nt < 2; ++nt)
      bfr[nt] = load_frag16(bufB + (wn + nt * 16) * LDSW, lane);

#pragma unroll
    for (int mt = 0; mt < 4; ++mt)
#pragma unroll
      for (int nt = 0; nt < 2; ++nt)
        acc[mt][nt] = __builtin_amdgcn_wmma_f32_16x16x32_f16(
            false, afr[mt], false, bfr[nt], (short)0, acc[mt][nt], false, false);

    buf = nbuf;
    nbuf = (nbuf == NBUF - 1) ? 0 : nbuf + 1;
  }

  // Epilogue: bias + ReLU, scatter back to original token rows (non-temporal:
  // keep L2 for the f16 operand tiles, the 128MB output is write-once).
  float bcol[2];
#pragma unroll
  for (int nt = 0; nt < 2; ++nt)
    bcol[nt] = bias[e * DOUT + n_block + wn + nt * 16 + (lane & 15)];

#pragma unroll
  for (int mt = 0; mt < 4; ++mt) {
    // C layout: VGPR j -> M=j (lanes 0-15) / M=j+8 (lanes 16-31); N = lane%16.
    int rbase = m_block + wm + mt * 16 + ((lane < 16) ? 0 : 8);
#pragma unroll
    for (int j = 0; j < 8; ++j) {
      int rl = rbase + j;
      if (rl < cnt) {
        int orig = row_map[seg + rl];
        float* orow = out + (size_t)orig * DOUT + n_block + wn + (lane & 15);
#pragma unroll
        for (int nt = 0; nt < 2; ++nt) {
          float v = acc[mt][nt][j] + bcol[nt];
          __builtin_nontemporal_store(v > 0.f ? v : 0.f, &orow[nt * 16]);
        }
      }
    }
  }
}

// ---------------- host-side launch ----------------
extern "C" void kernel_launch(void* const* d_in, const int* in_sizes, int n_in,
                              void* d_out, int out_size, void* d_ws, size_t ws_size,
                              hipStream_t stream) {
  (void)in_sizes; (void)n_in; (void)out_size; (void)ws_size;

  const float* x    = (const float*)d_in[0];
  const int*   idxs = (const int*)d_in[1];
  const float* Wf   = (const float*)d_in[2];
  const float* bias = (const float*)d_in[3];
  float* out = (float*)d_out;

  // Workspace layout (~134.3 MB):
  //   x_f16   : NTOK*DIN halfs            (64 MiB)
  //   W_f16   : NEXP*DOUT*DIN halfs       (64 MiB)
  //   row_map : NTOK ints                 (64 KiB)
  //   counts/offsets/cursors : 3*NEXP ints
  _Float16* xh = (_Float16*)d_ws;
  _Float16* wh = xh + (size_t)NTOK * DIN;
  int* row_map = (int*)(wh + (size_t)NEXP * DOUT * DIN);
  int* counts  = row_map + NTOK;
  int* offsets = counts + NEXP;
  int* cursors = offsets + NEXP;

  const int n4x = (NTOK * DIN) / 4;             // 8388608
  const int n4w = (NEXP * DOUT * DIN) / 4;      // 8388608
  k_cvt_f32_f16<<<(n4x + 255) / 256, 256, 0, stream>>>(x,  xh, n4x);
  k_cvt_f32_f16<<<(n4w + 255) / 256, 256, 0, stream>>>(Wf, wh, n4w);

  k_zero<<<1, 32, 0, stream>>>(counts, NEXP);
  k_hist<<<NTOK / 256, 256, 0, stream>>>(idxs, counts);
  k_scan<<<1, 1, 0, stream>>>(counts, offsets, cursors);
  k_assign<<<NTOK / 256, 256, 0, stream>>>(idxs, cursors, row_map);

  dim3 grid(DOUT / BN, NTOK / BM, NEXP);  // (16, 128, 8); blocks past segment exit early
  k_moe_gemm<<<grid, 256, 0, stream>>>(xh, wh, bias, row_map, counts, offsets, out);
}
